// SelfAttention_23562190586187
// MI455X (gfx1250) — compile-verified, matrix operand order checked by reading, not run on hardware
//
#include <hip/hip_runtime.h>

// ---------------------------------------------------------------------------
// Fused causal self-attention forward (B=2, N=2048, K=1024, H=16, S=64)
// bf16 WMMA everywhere; async global->LDS staging (ASYNCcnt) for GEMM operands.
// ---------------------------------------------------------------------------

typedef __bf16 bf16_t;
typedef __attribute__((ext_vector_type(16))) __bf16 v16bf;
typedef __attribute__((ext_vector_type(8)))  __bf16 v8bf;
typedef __attribute__((ext_vector_type(8)))  float  v8f;
typedef __attribute__((ext_vector_type(4)))  int    v4i;

#define WMMA_BF16(A, B, C) \
    __builtin_amdgcn_wmma_f32_16x16x32_bf16(false, (A), false, (B), (short)0, (C), false, false)

static constexpr int Bsz = 2;
static constexpr int Nsz = 2048;
static constexpr int Ksz = 1024;
static constexpr int Hsz = 16;
static constexpr int Ssz = 64;
static constexpr float LOG2E = 1.4426950408889634f;

__device__ __forceinline__ bf16_t f2bf(float f) {
    union { float f; unsigned u; } v; v.f = f;
    unsigned r = v.u + 0x7FFFu + ((v.u >> 16) & 1u);   // round-to-nearest-even
    unsigned short h = (unsigned short)(r >> 16);
    return __builtin_bit_cast(bf16_t, h);
}

// Async global->LDS 16B copy (ASYNCcnt path).
__device__ __forceinline__ void async_copy_b128(const bf16_t* g, bf16_t* l) {
#if __has_builtin(__builtin_amdgcn_global_load_async_to_lds_b128)
    __builtin_amdgcn_global_load_async_to_lds_b128(
        (__attribute__((address_space(1))) v4i*)g,
        (__attribute__((address_space(3))) v4i*)l, 0, 0);
#else
    unsigned loff = (unsigned)(uintptr_t)l;   // low 32 bits of flat LDS addr = LDS offset
    asm volatile("global_load_async_to_lds_b128 %0, %1, off" :: "v"(loff), "v"(g) : "memory");
#endif
}

__device__ __forceinline__ void wait_async0() {
#if __has_builtin(__builtin_amdgcn_s_wait_asynccnt)
    __builtin_amdgcn_s_wait_asynccnt(0);
#else
    asm volatile("s_wait_asynccnt 0x0" ::: "memory");
#endif
}

__device__ __forceinline__ void wait_ds0() {
    asm volatile("s_wait_dscnt 0x0" ::: "memory");
}

// Load one 16x32 bf16 WMMA fragment (A layout; identical for NT B fragments)
// from a row-major matrix (global or LDS): lane holds row (r0 + lane&15);
// element e -> k = k0 + (e&7) + ((e>>3)<<4) + ((lane>>4)<<3): two 16B chunks.
__device__ __forceinline__ v16bf load_frag(const bf16_t* base, int ld, int r0,
                                           int k0, int lane) {
    int r  = r0 + (lane & 15);
    int kb = k0 + ((lane >> 4) << 3);
    const bf16_t* p = base + (size_t)r * ld + kb;
    v8bf lo = *(const v8bf*)(p);
    v8bf hi = *(const v8bf*)(p + 16);
    v16bf a;
#pragma unroll
    for (int i = 0; i < 8; ++i) { a[i] = lo[i]; a[i + 8] = hi[i]; }
    return a;
}

// ---------------------------------------------------------------------------
__global__ void k_cast_bf16(const float* __restrict__ src, bf16_t* __restrict__ dst, int n8) {
    int i = blockIdx.x * blockDim.x + threadIdx.x;
    if (i < n8) {
        const float4* s4 = (const float4*)src + (size_t)i * 2;
        float4 a = s4[0], b = s4[1];
        v8bf o;
        o[0] = f2bf(a.x); o[1] = f2bf(a.y); o[2] = f2bf(a.z); o[3] = f2bf(a.w);
        o[4] = f2bf(b.x); o[5] = f2bf(b.y); o[6] = f2bf(b.z); o[7] = f2bf(b.w);
        *(v8bf*)(dst + (size_t)i * 8) = o;
    }
}

// ---------------------------------------------------------------------------
// Projection:  C(4096 x 1024) = Xbf @ Wbf^T.  One block (512 thr = 16 waves)
// per 16-row tile; the 16x1024 X strip is async-staged into LDS once and
// shared by all 16 waves (one head / 16x64 C tile per wave).
// mode 0/1: Q/K (+RoPE) -> dst[b][h][n][s];  mode 2: V -> dst[b][h][s][n].
// ---------------------------------------------------------------------------
__global__ __launch_bounds__(512) void k_proj(const bf16_t* __restrict__ X,
                                              const bf16_t* __restrict__ W,
                                              bf16_t* __restrict__ dst, int mode) {
    __shared__ bf16_t xs[16 * Ksz];               // 32 KB
    int tid  = threadIdx.x;
    int lane = tid & 31;
    int head = tid >> 5;                           // wave id = head, 0..15
    int row0 = blockIdx.x * 16;                    // 0..4080
    int col0 = head * Ssz;

    // Cooperative async stage: 16*1024 bf16 = 2048 x 16B chunks / 512 threads.
    const bf16_t* xsrc = X + (size_t)row0 * Ksz;
#pragma unroll
    for (int c = 0; c < 4; ++c)
        async_copy_b128(xsrc + (tid + 512 * c) * 8, xs + (tid + 512 * c) * 8);
    wait_async0();
    __syncthreads();

    v8f c[4] = {};
    for (int k0 = 0; k0 < Ksz; k0 += 32) {
        v16bf a = load_frag(xs, Ksz, 0, k0, lane);           // ds_load_b128
#pragma unroll
        for (int j = 0; j < 4; ++j) {
            v16bf b = load_frag(W, Ksz, col0 + 16 * j, k0, lane);
            c[j] = WMMA_BF16(a, b, c[j]);
        }
    }

    int half = lane >> 4;
    int nlo  = lane & 15;

    if (mode < 2) {  // RoPE: pairs (s, s+32) live in accumulators (j, j+2)
#pragma unroll
        for (int j = 0; j < 2; ++j) {
            int s = nlo + 16 * j;                             // 0..31
            float inv_freq = exp2f(-13.287712379549449f * (float)s * (1.0f / 32.0f));
#pragma unroll
            for (int r = 0; r < 8; ++r) {
                int m = row0 + r + 8 * half;
                float ang = (float)(m & (Nsz - 1)) * inv_freq;
                float cs = __cosf(ang), sn = __sinf(ang);
                float lo = c[j][r], hi = c[j + 2][r];
                c[j][r]     = lo * cs - hi * sn;
                c[j + 2][r] = hi * cs + lo * sn;
            }
        }
    }

#pragma unroll
    for (int j = 0; j < 4; ++j)
#pragma unroll
        for (int r = 0; r < 8; ++r) {
            int m = row0 + r + 8 * half;                      // [0, B*N)
            int b = m >> 11;
            int n = m & (Nsz - 1);
            int s = nlo + 16 * j;
            bf16_t v = f2bf(c[j][r]);
            if (mode == 2)
                dst[(((size_t)(b * Hsz + head)) * Ssz + s) * Nsz + n] = v;  // Vt
            else
                dst[(((size_t)(b * Hsz + head)) * Nsz + n) * Ssz + s] = v;  // Q/K
        }
}

// ---------------------------------------------------------------------------
// Flash attention: one block (128 thr = 4 waves) per (b,h, 64-row q chunk).
// Key loop bound is block-uniform so K/V tiles are async-staged into LDS once
// per 32-key step and shared by all 4 waves; fully-masked blocks for the lower
// waves contribute exact zeros through the online softmax.
// ---------------------------------------------------------------------------
__global__ __launch_bounds__(128) void k_attn(const bf16_t* __restrict__ Q,
                                              const bf16_t* __restrict__ Kh,
                                              const bf16_t* __restrict__ Vt,
                                              bf16_t* __restrict__ O) {
    __shared__ bf16_t kt[32 * Ssz];        // K tile  [32 keys][64]   4 KB
    __shared__ bf16_t vts[Ssz * 32];       // Vt tile [64 s][32 keys] 4 KB
    __shared__ bf16_t pbuf[4 * 16 * 32];   // per-wave P tiles        4 KB

    int tid  = threadIdx.x;
    int lane = tid & 31;
    int wave = tid >> 5;
    int bh   = blockIdx.x >> 5;            // (b*16 + h), 32 q-chunks per bh
    int q0b  = (blockIdx.x & 31) * 64;     // chunk base
    int q0   = q0b + wave * 16;            // this wave's q tile

    const bf16_t* Qb = Q  + (size_t)bh * Nsz * Ssz;
    const bf16_t* Kb = Kh + (size_t)bh * Nsz * Ssz;
    const bf16_t* Vb = Vt + (size_t)bh * Ssz * Nsz;

    v16bf qa0 = load_frag(Qb, Ssz, q0, 0, lane);
    v16bf qa1 = load_frag(Qb, Ssz, q0, 32, lane);

    v8f o[4] = {};
    float M[8], L[8];
#pragma unroll
    for (int r = 0; r < 8; ++r) { M[r] = -3.0e38f; L[r] = 0.0f; }

    int half = lane >> 4;
    int nlo  = lane & 15;
    bf16_t* pl = pbuf + wave * (16 * 32);

    for (int kb = 0; kb < q0b + 64; kb += 32) {        // uniform across block
        // ---- cooperative async stage of K (4KB contiguous) and Vt (64 x 64B)
        {
            const bf16_t* ks = Kb + (size_t)kb * Ssz;  // rows kb..kb+31, contiguous
            async_copy_b128(ks + tid * 8, kt + tid * 8);
            async_copy_b128(ks + (tid + 128) * 8, kt + (tid + 128) * 8);
            int j0 = tid, j1 = tid + 128;              // 256 chunks: s=j>>2, c=j&3
            async_copy_b128(Vb + (size_t)(j0 >> 2) * Nsz + kb + (j0 & 3) * 8,
                            vts + (j0 >> 2) * 32 + (j0 & 3) * 8);
            async_copy_b128(Vb + (size_t)(j1 >> 2) * Nsz + kb + (j1 & 3) * 8,
                            vts + (j1 >> 2) * 32 + (j1 & 3) * 8);
        }
        wait_async0();
        __syncthreads();

        // ---- scores: 16(q) x 32(keys), K-dim = S = 64 -> 2 WMMAs per 16 keys
        v8f s[2] = {};
#pragma unroll
        for (int jn = 0; jn < 2; ++jn) {
            v16bf kf0 = load_frag(kt, Ssz, 16 * jn, 0, lane);
            v16bf kf1 = load_frag(kt, Ssz, 16 * jn, 32, lane);
            s[jn] = WMMA_BF16(qa0, kf0, s[jn]);
            s[jn] = WMMA_BF16(qa1, kf1, s[jn]);
        }
        // ---- scale + causal mask
#pragma unroll
        for (int jn = 0; jn < 2; ++jn) {
            int kidx = kb + nlo + 16 * jn;
#pragma unroll
            for (int r = 0; r < 8; ++r) {
                int qi = q0 + r + 8 * half;
                float v = s[jn][r] * 0.125f;          // 1/sqrt(64)
                s[jn][r] = (kidx <= qi) ? v : -3.0e38f;
            }
        }
        // ---- online softmax row stats (row r lives across a 16-lane half)
        float A[8];
#pragma unroll
        for (int r = 0; r < 8; ++r) {
            float bm = fmaxf(s[0][r], s[1][r]);
#pragma unroll
            for (int d = 1; d < 16; d <<= 1) bm = fmaxf(bm, __shfl_xor(bm, d, 32));
            float Mn = fmaxf(M[r], bm);
            A[r] = exp2f((M[r] - Mn) * LOG2E);
            float p0 = exp2f((s[0][r] - Mn) * LOG2E);
            float p1 = exp2f((s[1][r] - Mn) * LOG2E);
            s[0][r] = p0; s[1][r] = p1;
            float ps = p0 + p1;
#pragma unroll
            for (int d = 1; d < 16; d <<= 1) ps += __shfl_xor(ps, d, 32);
            L[r] = L[r] * A[r] + ps;
            M[r] = Mn;
        }
#pragma unroll
        for (int j = 0; j < 4; ++j)
#pragma unroll
            for (int r = 0; r < 8; ++r) o[j][r] *= A[r];

        // ---- P (C layout) -> per-wave LDS -> A-layout fragment
#pragma unroll
        for (int jn = 0; jn < 2; ++jn)
#pragma unroll
            for (int r = 0; r < 8; ++r)
                pl[(r + 8 * half) * 32 + nlo + 16 * jn] = f2bf(s[jn][r]);
        wait_ds0();
        v16bf pf = load_frag(pl, 32, 0, 0, lane);

        // ---- O += P(16x32) @ V(32x64), V fragments from staged Vt tile
#pragma unroll
        for (int j = 0; j < 4; ++j) {
            v16bf vf = load_frag(vts, 32, 16 * j, 0, lane);
            o[j] = WMMA_BF16(pf, vf, o[j]);
        }
        __syncthreads();   // tiles consumed; safe to overwrite next iteration
    }

    int b = bh >> 4, h = bh & 15;
#pragma unroll
    for (int r = 0; r < 8; ++r) {
        float rl = 1.0f / L[r];
#pragma unroll
        for (int j = 0; j < 4; ++j) {
            int n = q0 + r + 8 * half;
            int col = h * Ssz + 16 * j + nlo;
            O[((size_t)(b * Nsz + n)) * Ksz + col] = f2bf(o[j][r] * rl);
        }
    }
}

// ---------------------------------------------------------------------------
// Output projection: out = attn(4096x1024) @ Wu^T + bu (fp32 out).
// Same LDS-staged structure as k_proj; wave w computes cols [64w, 64w+64).
// ---------------------------------------------------------------------------
__global__ __launch_bounds__(512) void k_outproj(const bf16_t* __restrict__ A,
                                                 const bf16_t* __restrict__ W,
                                                 const float* __restrict__ bias,
                                                 float* __restrict__ out) {
    __shared__ bf16_t xs[16 * Ksz];               // 32 KB
    int tid  = threadIdx.x;
    int lane = tid & 31;
    int wave = tid >> 5;
    int row0 = blockIdx.x * 16;
    int col0 = wave * 64;

    const bf16_t* asrc = A + (size_t)row0 * Ksz;
#pragma unroll
    for (int cc = 0; cc < 4; ++cc)
        async_copy_b128(asrc + (tid + 512 * cc) * 8, xs + (tid + 512 * cc) * 8);
    wait_async0();
    __syncthreads();

    v8f c[4] = {};
    for (int k0 = 0; k0 < Ksz; k0 += 32) {
        v16bf a = load_frag(xs, Ksz, 0, k0, lane);
#pragma unroll
        for (int j = 0; j < 4; ++j) {
            v16bf b = load_frag(W, Ksz, col0 + 16 * j, k0, lane);
            c[j] = WMMA_BF16(a, b, c[j]);
        }
    }
    int half = lane >> 4, nlo = lane & 15;
#pragma unroll
    for (int j = 0; j < 4; ++j)
#pragma unroll
        for (int r = 0; r < 8; ++r) {
            int m = row0 + r + 8 * half;
            int col = col0 + 16 * j + nlo;
            out[(size_t)m * Ksz + col] = c[j][r] + bias[col];
        }
}

// ---------------------------------------------------------------------------
extern "C" void kernel_launch(void* const* d_in, const int* in_sizes, int n_in,
                              void* d_out, int out_size, void* d_ws, size_t ws_size,
                              hipStream_t stream) {
    const float* x  = (const float*)d_in[0];
    const float* Wq = (const float*)d_in[1];
    const float* Wk = (const float*)d_in[2];
    const float* Wv = (const float*)d_in[3];
    const float* Wu = (const float*)d_in[4];
    const float* bu = (const float*)d_in[5];
    // d_in[6] = mask: causal, implemented analytically.
    float* out = (float*)d_out;

    const size_t NX = (size_t)Bsz * Nsz * Ksz;   // 4 Mi elements
    const size_t NW = (size_t)Ksz * Ksz;         // 1 Mi elements
    char* ws = (char*)d_ws;
    size_t off = 0;
    bf16_t* xbf  = (bf16_t*)(ws + off); off += NX * 2;
    bf16_t* wqbf = (bf16_t*)(ws + off); off += NW * 2;
    bf16_t* wkbf = (bf16_t*)(ws + off); off += NW * 2;
    bf16_t* wvbf = (bf16_t*)(ws + off); off += NW * 2;
    bf16_t* wubf = (bf16_t*)(ws + off); off += NW * 2;
    bf16_t* Qb   = (bf16_t*)(ws + off); off += NX * 2;
    bf16_t* Kb   = (bf16_t*)(ws + off); off += NX * 2;
    bf16_t* Vtb  = (bf16_t*)(ws + off); off += NX * 2;
    bf16_t* AOb  = (bf16_t*)(ws + off); off += NX * 2;   // total 48 MiB

    k_cast_bf16<<<dim3((NX / 8 + 255) / 256), 256, 0, stream>>>(x,  xbf,  (int)(NX / 8));
    k_cast_bf16<<<dim3((NW / 8 + 255) / 256), 256, 0, stream>>>(Wq, wqbf, (int)(NW / 8));
    k_cast_bf16<<<dim3((NW / 8 + 255) / 256), 256, 0, stream>>>(Wk, wkbf, (int)(NW / 8));
    k_cast_bf16<<<dim3((NW / 8 + 255) / 256), 256, 0, stream>>>(Wv, wvbf, (int)(NW / 8));
    k_cast_bf16<<<dim3((NW / 8 + 255) / 256), 256, 0, stream>>>(Wu, wubf, (int)(NW / 8));

    dim3 pgrid((Bsz * Nsz) / 16);   // 256 blocks x 512 threads (16 waves = 16 heads)
    k_proj<<<pgrid, 512, 0, stream>>>(xbf, wqbf, Qb, 0);
    k_proj<<<pgrid, 512, 0, stream>>>(xbf, wkbf, Kb, 1);
    k_proj<<<pgrid, 512, 0, stream>>>(xbf, wvbf, Vtb, 2);

    k_attn<<<dim3(Bsz * Hsz * (Nsz / 64)), 128, 0, stream>>>(Qb, Kb, Vtb, AOb);

    k_outproj<<<pgrid, 512, 0, stream>>>(AOb, wubf, bu, out);
}
